// AdvancedRNN_28613072126749
// MI455X (gfx1250) — matrix operand order repaired and checked by reading, not compiled, not run on hardware
//
#include <hip/hip_runtime.h>

// ---------------------------------------------------------------------------
// AdvancedRNN on MI455X (gfx1250): persistent per-batch-tile RNN with
// bf16x3-split WMMA (v_wmma_f32_16x16x32_bf16) for the 512x512 recurrent matmul.
// B fragments pre-swizzled into WMMA layout with hi/lo interleaved at 64B per
// fragment-lane so the inner loop needs one 32-bit voffset per (kc, nb).
// ---------------------------------------------------------------------------

typedef __attribute__((ext_vector_type(16))) __bf16          v16bf;
typedef __attribute__((ext_vector_type(8)))  float           v8f;
typedef __attribute__((ext_vector_type(16))) unsigned short  v16u;

#define RNN_B   1024
#define RNN_H   512
#define RNN_T   256
#define RNN_NS  64
#define HSTR    516   // padded LDS row stride (floats) to spread banks

__device__ __forceinline__ unsigned short f32_to_bf16_rne(float f) {
  unsigned u = __float_as_uint(f);
  u += 0x7FFFu + ((u >> 16) & 1u);
  return (unsigned short)(u >> 16);
}
__device__ __forceinline__ float bf16u_to_f32(unsigned short s) {
  return __uint_as_float(((unsigned)s) << 16);
}

// ---------------------------------------------------------------------------
// Prep: split W_hh into bf16 hi/lo in WMMA B-fragment layout, interleaved.
// B matrix is K x N with B[k][n] = W_hh[n][k].
// Fragment (kc, nb), lane = (klocal/16)*16 + (n%16):
//   32 ushorts at fbase = ((kc*32 + nb)*32 + lane)*32
//   [0..15] = hi (e = k%16), [16..31] = lo
// ---------------------------------------------------------------------------
__global__ void rnn_prep_kernel(const float* __restrict__ Whh,
                                unsigned short* __restrict__ wfrag) {
  int idx = blockIdx.x * blockDim.x + threadIdx.x;
  if (idx >= RNN_H * RNN_H) return;
  int n = idx >> 9;          // row of W_hh (output neuron)
  int k = idx & (RNN_H - 1); // input index
  float w = Whh[n * RNN_H + k];
  unsigned short hb = f32_to_bf16_rne(w);
  unsigned short lb = f32_to_bf16_rne(w - bf16u_to_f32(hb));
  int kc    = k >> 5;
  int nb    = n >> 4;
  int half  = (k >> 4) & 1;
  int e     = k & 15;
  int lane  = half * 16 + (n & 15);
  int fbase = (((kc * 32 + nb) * 32) + lane) * 32;
  wfrag[fbase + e]      = hb;
  wfrag[fbase + 16 + e] = lb;
}

// ---------------------------------------------------------------------------
// Main persistent kernel: 64 blocks x 512 threads (16 waves of 32).
// Each block owns 16 batch rows; h (16x512 fp32) lives in LDS.
// Per step: build split-A fragments cooperatively, then each wave does the
// GEMM for its 2 N-blocks (6 wmma per k-chunk: hi*hi, lo*hi, hi*lo).
// ---------------------------------------------------------------------------
__global__ __launch_bounds__(512, 1)
void rnn_persist_kernel(const float* __restrict__ x,
                        const float* __restrict__ Wih,
                        const float* __restrict__ bih,
                        const float* __restrict__ bhh,
                        const float* __restrict__ Wfc,
                        const float* __restrict__ bfc,
                        const unsigned short* __restrict__ wfrag,
                        float* __restrict__ out) {
  __shared__ __align__(16) float hS[16 * HSTR];               // fp32 hidden state
  __shared__ __align__(32) unsigned short AS[16 * 32 * 32];   // A-frag hi|lo interleaved
  __shared__ float biasS[RNN_H];
  __shared__ float winS[RNN_H];
  __shared__ float wfcS[RNN_H];
  __shared__ float sv[16];                                    // per-row scalar drive

  const int tid  = threadIdx.x;
  const int wave = tid >> 5;
  const int lane = tid & 31;
  const int nl   = lane & 15;
  const int hf   = lane >> 4;
  const int row0 = blockIdx.x * 16;

  // one-time per-block init
  biasS[tid] = bih[tid] + bhh[tid];
  winS[tid]  = Wih[tid];   // W_ih is (H,1): column vector
  wfcS[tid]  = Wfc[tid];   // W_fc is (1,H): row vector
  for (int i = tid; i < 16 * HSTR; i += 512) hS[i] = 0.f;
  __syncthreads();

  const int   nb0 = wave * 2, nb1 = wave * 2 + 1;
  const float wv0 = winS[nb0 * 16 + nl], wv1 = winS[nb1 * 16 + nl];
  const float bv0 = biasS[nb0 * 16 + nl], bv1 = biasS[nb1 * 16 + nl];
  const float bfc0 = bfc[0];

  // byte offsets of this wave's B fragments (kc advances by 64 KB)
  const char*    wb    = (const char*)wfrag;
  const unsigned boff0 = (unsigned)nb0 * 2048u + (unsigned)lane * 64u;
  const unsigned boff1 = (unsigned)nb1 * 2048u + (unsigned)lane * 64u;
  // this lane's A-fragment base in LDS (kc advances by 2 KB)
  const unsigned short* ap0 = &AS[(unsigned)lane * 32u];

  for (int t = 0; t < RNN_T + RNN_NS; ++t) {
    // ---- phase 1: scalar drive s[m] ----
    if (t < RNN_T) {
      if (tid < 16) sv[tid] = x[(row0 + tid) * RNN_T + t];
    } else {
      // rollout: y[m] = h[m,:] . W_fc + b_fc   (wave w handles row m = w)
      const int m = wave;
      const float* hr = &hS[m * HSTR];
      float p = 0.f;
      #pragma unroll
      for (int j = 0; j < 16; ++j) p += hr[lane + 32 * j] * wfcS[lane + 32 * j];
      #pragma unroll
      for (int off = 16; off >= 1; off >>= 1) p += __shfl_xor(p, off, 32);
      if (lane == 0) {
        const float y = p + bfc0;
        sv[m] = y;
        out[(row0 + m) * RNN_NS + (t - RNN_T)] = y;  // (B, NUM_STEPS, 1)
      }
    }
    __syncthreads();

    // ---- phase 2: split h -> A fragments (wave w builds k-chunk kc = w) ----
    {
      const int kc = wave;
      const int m  = nl;
      const float* hp = &hS[m * HSTR + kc * 32 + hf * 16];
      float a[16];
      const float4 f0 = *(const float4*)(hp);
      const float4 f1 = *(const float4*)(hp + 4);
      const float4 f2 = *(const float4*)(hp + 8);
      const float4 f3 = *(const float4*)(hp + 12);
      a[0] = f0.x;  a[1] = f0.y;  a[2] = f0.z;  a[3] = f0.w;
      a[4] = f1.x;  a[5] = f1.y;  a[6] = f1.z;  a[7] = f1.w;
      a[8] = f2.x;  a[9] = f2.y;  a[10] = f2.z; a[11] = f2.w;
      a[12] = f3.x; a[13] = f3.y; a[14] = f3.z; a[15] = f3.w;
      v16u uh, ul;
      #pragma unroll
      for (int e = 0; e < 16; ++e) {
        const unsigned short hb = f32_to_bf16_rne(a[e]);
        uh[e] = hb;
        ul[e] = f32_to_bf16_rne(a[e] - bf16u_to_f32(hb));
      }
      unsigned short* as = &AS[((unsigned)kc * 32u + (unsigned)lane) * 32u];
      *(v16u*)(as)      = uh;
      *(v16u*)(as + 16) = ul;
    }
    __syncthreads();

    // ---- phase 3: GEMM h @ W_hh^T via bf16x3 WMMA ----
    v8f acc0 = {0.f, 0.f, 0.f, 0.f, 0.f, 0.f, 0.f, 0.f};
    v8f acc1 = {0.f, 0.f, 0.f, 0.f, 0.f, 0.f, 0.f, 0.f};
    unsigned o0 = boff0, o1 = boff1;
    const unsigned short* ap = ap0;
    #pragma unroll 4
    for (int kc = 0; kc < 16; ++kc) {
      const v16bf ah  = *(const v16bf*)(ap);
      const v16bf al  = *(const v16bf*)(ap + 16);
      const v16bf bh0 = *(const v16bf*)(wb + o0);
      const v16bf bl0 = *(const v16bf*)(wb + o0 + 32);
      const v16bf bh1 = *(const v16bf*)(wb + o1);
      const v16bf bl1 = *(const v16bf*)(wb + o1 + 32);
      acc0 = __builtin_amdgcn_wmma_f32_16x16x32_bf16(false, ah, false, bh0, (short)0, acc0, false, false);
      acc1 = __builtin_amdgcn_wmma_f32_16x16x32_bf16(false, ah, false, bh1, (short)0, acc1, false, false);
      acc0 = __builtin_amdgcn_wmma_f32_16x16x32_bf16(false, al, false, bh0, (short)0, acc0, false, false);
      acc1 = __builtin_amdgcn_wmma_f32_16x16x32_bf16(false, al, false, bh1, (short)0, acc1, false, false);
      acc0 = __builtin_amdgcn_wmma_f32_16x16x32_bf16(false, ah, false, bl0, (short)0, acc0, false, false);
      acc1 = __builtin_amdgcn_wmma_f32_16x16x32_bf16(false, ah, false, bl1, (short)0, acc1, false, false);
      o0 += 65536u;     // next kc: 32 nb * 32 lanes * 64 B
      o1 += 65536u;
      ap += 32u * 32u;  // next kc in LDS (2 KB of ushorts)
    }

    // ---- phase 4: epilogue  h = relu(acc + s*w_in + bias) ----
    float svr[8];
    #pragma unroll
    for (int r = 0; r < 8; ++r) svr[r] = sv[r + 8 * hf];
    #pragma unroll
    for (int r = 0; r < 8; ++r) {
      const int m = r + 8 * hf;           // C/D layout: row = r + 8*(lane/16)
      float v0 = acc0[r] + svr[r] * wv0 + bv0;
      float v1 = acc1[r] + svr[r] * wv1 + bv1;
      v0 = fmaxf(v0, 0.f);
      v1 = fmaxf(v1, 0.f);
      hS[m * HSTR + nb0 * 16 + nl] = v0;
      hS[m * HSTR + nb1 * 16 + nl] = v1;
    }
    __syncthreads();
  }
}

// ---------------------------------------------------------------------------
// Launcher
// ---------------------------------------------------------------------------
extern "C" void kernel_launch(void* const* d_in, const int* in_sizes, int n_in,
                              void* d_out, int out_size, void* d_ws, size_t ws_size,
                              hipStream_t stream) {
  const float* x    = (const float*)d_in[0];
  const float* Wih  = (const float*)d_in[1];
  const float* Whh  = (const float*)d_in[2];
  const float* bih  = (const float*)d_in[3];
  const float* bhh  = (const float*)d_in[4];
  const float* Wfc  = (const float*)d_in[5];
  const float* bfc  = (const float*)d_in[6];
  (void)in_sizes; (void)n_in; (void)out_size; (void)ws_size;

  unsigned short* wfrag = (unsigned short*)d_ws;   // 1 MB: hi|lo interleaved

  rnn_prep_kernel<<<dim3((RNN_H * RNN_H + 255) / 256), dim3(256), 0, stream>>>(Whh, wfrag);
  rnn_persist_kernel<<<dim3(RNN_B / 16), dim3(512), 0, stream>>>(
      x, Wih, bih, bhh, Wfc, bfc, wfrag, (float*)d_out);
}
